// VectorQuantizerEMAKeras_70669391889038
// MI455X (gfx1250) — compile-verified, matrix operand order checked by reading, not run on hardware
//
#include <hip/hip_runtime.h>
#include <hip/hip_bf16.h>

typedef __attribute__((ext_vector_type(16))) __bf16 v16bf;
typedef __attribute__((ext_vector_type(8)))  float  v8f;
typedef __attribute__((ext_vector_type(4)))  unsigned int u32x4;
typedef __attribute__((ext_vector_type(8)))  int i32x8;
typedef __attribute__((ext_vector_type(4)))  int i32x4;

#define DIM     64
#define KCODES  512
#define N_ROWS  131072          // 32*64*64
#define NUMEL   (N_ROWS * DIM)  // 8388608

// d_out float offsets
#define OUT_Q     0
#define OUT_LOSS  8388608
#define OUT_PERP  8388609
#define OUT_NEMB  8388610
#define OUT_IDX   8421378

// d_ws float offsets
#define WS_LOSS    0        // 1
#define WS_CLUSTER 1        // 512
#define WS_DW      1024     // 64*512 = 32768
#define WS_ESQ     33792    // 512
#define WS_EMBT    34304    // 512*64 = 32768 (codebook transposed, [K][D] fp32)
#define WS_FRAG    67072    // 16384 floats = 32768 bf16 (B fragments)

// TDM path: 5-arg builtin on ROCm 7.2 (no therock TDM header); fall back to
// manual copy on the 6-arg toolchain or if the builtin is missing.
#if defined(__has_builtin)
#if __has_builtin(__builtin_amdgcn_tensor_load_to_lds) && \
    __has_builtin(__builtin_amdgcn_s_wait_tensorcnt) && \
    !__has_include(<hip/amd_detail/amd_gfx1250_TDM.h>)
#define USE_TDM 1
#endif
#endif
#ifndef USE_TDM
#define USE_TDM 0
#endif

// ---------------------------------------------------------------------------
// Prep: zero accumulators, ||e||^2, transposed codebook, bf16 B-fragments
// packed in the 16-bit WMMA B layout (ISA 7.12.2).
// Fragment slot s = ((t*2 + kk)*32 + lane)*16 + i
//   h = lane>>4, n = lane&15
//   k_local = (i<8) ? 8h+i : 16+8h+(i-8);  d = 32*kk + k_local;  c = 16*t + n
// ---------------------------------------------------------------------------
__global__ void vq_prep(const float* __restrict__ emb, float* __restrict__ ws) {
    int tid = blockIdx.x * blockDim.x + threadIdx.x;
    if (tid < 33792) ws[tid] = 0.0f;                       // loss + cluster + dw (+pad)
    if (tid < KCODES * DIM) {                              // embT[k][d] = emb[d][k]
        int k = tid >> 6, d = tid & 63;
        ws[WS_EMBT + tid] = emb[d * KCODES + k];
    }
    if (tid < KCODES * DIM) {                              // bf16 B fragments
        int s    = tid;
        int t    = s >> 10;
        int r    = s & 1023;
        int kk   = r >> 9;
        int r2   = r & 511;
        int lane = r2 >> 4;
        int i    = r2 & 15;
        int h    = lane >> 4;
        int n    = lane & 15;
        int k_local = (i < 8) ? (8 * h + i) : (16 + 8 * h + (i - 8));
        int d = 32 * kk + k_local;
        int c = 16 * t + n;
        __bf16* frag = (__bf16*)(ws + WS_FRAG);
        frag[s] = (__bf16)emb[d * KCODES + c];
    }
    if (tid < KCODES) {                                    // ||e_k||^2
        float s = 0.0f;
        #pragma unroll
        for (int d = 0; d < DIM; ++d) {
            float v = emb[d * KCODES + tid];
            s += v * v;
        }
        ws[WS_ESQ + tid] = s;
    }
}

// ---------------------------------------------------------------------------
// Main: distance GEMM via v_wmma_f32_16x16x32_bf16 (two 16-row tiles per
// wave, B double-buffered from LDS with peeled last iteration), argmin,
// quantize, EMA statistics.
// ---------------------------------------------------------------------------
__global__ void __launch_bounds__(256) vq_main(const float* __restrict__ x,
                                               float* __restrict__ out,
                                               float* __restrict__ ws) {
    __shared__ __align__(32) __bf16 sB[32 * 2 * 32 * 16];  // 64 KB B fragments
    __shared__ float sEsq[KCODES];

    const int tid  = threadIdx.x;
    const int lane = tid & 31;
    const int wave = tid >> 5;

    // ---- Stage codebook fragments into LDS ----
#if USE_TDM
    if (wave == 0) {
        unsigned lds_addr        = (unsigned)(unsigned long long)(void*)sB;
        unsigned long long gaddr = (unsigned long long)(const void*)(ws + WS_FRAG);
        u32x4 g0;
        g0[0] = 1u;                                        // count=1 user descriptor
        g0[1] = lds_addr;                                  // lds_addr
        g0[2] = (unsigned)(gaddr & 0xFFFFFFFFu);           // global_addr[31:0]
        g0[3] = (unsigned)((gaddr >> 32) & 0x1FFFFFFu)     // global_addr[56:32]
                | (2u << 30);                              // type=2 (image)
        i32x8 g1;
        g1[0] = (int)(2u << 16);          // data_size=4B; no mask/iterate/pad
        g1[1] = (int)(16384u << 16);      // tensor_dim0 = 16384 (low 16 -> [31:16])
        g1[2] = (int)(1u << 16);          // tensor_dim0 hi = 0 | tensor_dim1 = 1
        g1[3] = (int)(16384u << 16);      // tile_dim0 = 16384
        g1[4] = 1;                        // tile_dim1 = 1, tile_dim2 = 0
        g1[5] = 16384;                    // tensor_dim0_stride low
        g1[6] = 0;                        // stride0 hi | stride1 lo
        g1[7] = 0;
        i32x4 gz = {0, 0, 0, 0};          // groups 2/3 unused (2D tensor)
        __builtin_amdgcn_tensor_load_to_lds(g0, g1, gz, gz, 0);
        __builtin_amdgcn_s_wait_tensorcnt(0);
    }
#else
    {
        const uint4* src = (const uint4*)(ws + WS_FRAG);
        uint4*       dst = (uint4*)sB;
        for (int j = tid; j < 4096; j += 256) dst[j] = src[j];
    }
#endif
    for (int j = tid; j < KCODES; j += 256) sEsq[j] = ws[WS_ESQ + j];
    __syncthreads();

    const int h   = lane >> 4;
    const int m16 = lane & 15;

    float*       cluster = ws + WS_CLUSTER;
    float*       dw      = ws + WS_DW;
    const float* embT    = ws + WS_EMBT;
    float*       q_out   = out + OUT_Q;
    float*       idx_out = out + OUT_IDX;

    float lsum = 0.0f;
    const int npairs = N_ROWS / 32;      // dual 16-row tiles per wave

    for (int pt = blockIdx.x * 8 + wave; pt < npairs; pt += gridDim.x * 8) {
        const int rowbase = pt * 32;

        // A fragments: two 16x32 tiles x two K-halves (ISA 16-bit A layout)
        v16bf a[2][2];
        #pragma unroll
        for (int u = 0; u < 2; ++u) {
            #pragma unroll
            for (int kk = 0; kk < 2; ++kk) {
                const float* rp = x + (size_t)(rowbase + 16 * u + m16) * DIM
                                  + 32 * kk + 8 * h;
                float4 f0 = *(const float4*)(rp + 0);
                float4 f1 = *(const float4*)(rp + 4);
                float4 f2 = *(const float4*)(rp + 16);
                float4 f3 = *(const float4*)(rp + 20);
                v16bf av;
                av[0]  = (__bf16)f0.x; av[1]  = (__bf16)f0.y; av[2]  = (__bf16)f0.z; av[3]  = (__bf16)f0.w;
                av[4]  = (__bf16)f1.x; av[5]  = (__bf16)f1.y; av[6]  = (__bf16)f1.z; av[7]  = (__bf16)f1.w;
                av[8]  = (__bf16)f2.x; av[9]  = (__bf16)f2.y; av[10] = (__bf16)f2.z; av[11] = (__bf16)f2.w;
                av[12] = (__bf16)f3.x; av[13] = (__bf16)f3.y; av[14] = (__bf16)f3.z; av[15] = (__bf16)f3.w;
                a[u][kk] = av;
            }
        }

        float bestv[2][8];
        int   besti[2][8];
        #pragma unroll
        for (int u = 0; u < 2; ++u)
            #pragma unroll
            for (int r = 0; r < 8; ++r) { bestv[u][r] = 3.4e38f; besti[u][r] = 0; }

        // One column-tile step: 4 WMMAs in two accumulator chains + packed score/argmin
        auto step = [&](int t, v16bf b0, v16bf b1) {
            v8f c0 = {};
            v8f c1 = {};
            c0 = __builtin_amdgcn_wmma_f32_16x16x32_bf16(false, a[0][0], false, b0,
                                                         (short)0, c0, false, false);
            c1 = __builtin_amdgcn_wmma_f32_16x16x32_bf16(false, a[1][0], false, b0,
                                                         (short)0, c1, false, false);
            c0 = __builtin_amdgcn_wmma_f32_16x16x32_bf16(false, a[0][1], false, b1,
                                                         (short)0, c0, false, false);
            c1 = __builtin_amdgcn_wmma_f32_16x16x32_bf16(false, a[1][1], false, b1,
                                                         (short)0, c1, false, false);
            const int   code = t * 16 + m16;
            const float e2   = sEsq[code];
            #pragma unroll
            for (int r = 0; r < 8; ++r) {
                float s0 = e2 - 2.0f * c0[r];   // ||x||^2 constant per row: dropped
                float s1 = e2 - 2.0f * c1[r];
                if (s0 < bestv[0][r]) { bestv[0][r] = s0; besti[0][r] = code; }
                if (s1 < bestv[1][r]) { bestv[1][r] = s1; besti[1][r] = code; }
            }
        };

        // Pipelined B reads: steady-state loop prefetches t+1 unconditionally,
        // last tile peeled (no per-iteration branch in the hot loop).
        v16bf b0 = *(const v16bf*)(sB + ((0 * 2 + 0) * 32 + lane) * 16);
        v16bf b1 = *(const v16bf*)(sB + ((0 * 2 + 1) * 32 + lane) * 16);
        for (int t = 0; t < 31; ++t) {
            v16bf nb0 = *(const v16bf*)(sB + (((t + 1) * 2 + 0) * 32 + lane) * 16);
            v16bf nb1 = *(const v16bf*)(sB + (((t + 1) * 2 + 1) * 32 + lane) * 16);
            step(t, b0, b1);
            b0 = nb0;
            b1 = nb1;
        }
        step(31, b0, b1);

        // Min-reduce across the 16 lanes holding each row (xor stays in half)
        #pragma unroll
        for (int mask = 1; mask < 16; mask <<= 1) {
            #pragma unroll
            for (int u = 0; u < 2; ++u) {
                #pragma unroll
                for (int r = 0; r < 8; ++r) {
                    float ov = __shfl_xor(bestv[u][r], mask, 32);
                    int   oi = __shfl_xor(besti[u][r], mask, 32);
                    if (ov < bestv[u][r] ||
                        (ov == bestv[u][r] && oi < besti[u][r])) {
                        bestv[u][r] = ov; besti[u][r] = oi;
                    }
                }
            }
        }

        // Per-row epilogue: index, quantized row (coalesced), loss, EMA stats
        #pragma unroll
        for (int u = 0; u < 2; ++u) {
            #pragma unroll
            for (int m = 0; m < 16; ++m) {
                const int r       = m & 7;
                const int srcLane = (m >> 3) << 4;        // 0 or 16
                const int idx     = __shfl(besti[u][r], srcLane, 32);
                const int row     = rowbase + 16 * u + m;
                if (lane == 0) {
                    atomicAdd(cluster + idx, 1.0f);
                    idx_out[row] = (float)idx;
                }
                const float* ecol = embT + (size_t)idx * DIM;
                const float* xr   = x + (size_t)row * DIM;
                float*       qr   = q_out + (size_t)row * DIM;
                #pragma unroll
                for (int cc = 0; cc < 2; ++cc) {
                    const int col = lane + 32 * cc;
                    float q  = ecol[col];
                    float xv = xr[col];
                    qr[col] = q;                          // quantized_st == q in value
                    float df = q - xv;
                    lsum += df * df;
                    atomicAdd(dw + col * KCODES + idx, xv);
                }
            }
        }
    }

    // loss partial: wave reduce, one atomic per wave
    #pragma unroll
    for (int mask = 1; mask < 32; mask <<= 1) lsum += __shfl_xor(lsum, mask, 32);
    if (lane == 0) atomicAdd(ws + WS_LOSS, lsum);
}

// ---------------------------------------------------------------------------
// Finalize: EMA debias, Laplace smoothing, new codebook, loss, perplexity
// ---------------------------------------------------------------------------
__global__ void __launch_bounds__(512) vq_fin(const float* __restrict__ ema_c,
                                              const float* __restrict__ ema_dw,
                                              const int*   __restrict__ counter,
                                              float* __restrict__ out,
                                              float* __restrict__ ws) {
    __shared__ float red[KCODES];
    const int k = threadIdx.x;

    const float t      = (float)(counter[0] + 1);
    const float debias = 1.0f - __powf(0.99f, t);

    const float cn    = ws[WS_CLUSTER + k];
    const float nh    = ema_c[k] - (ema_c[k] - cn) * 0.01f;   // 1 - DECAY = 0.01
    const float avg_c = nh / debias;

    // n_total = sum(avg_c)
    red[k] = avg_c;
    __syncthreads();
    for (int s = 256; s > 0; s >>= 1) {
        if (k < s) red[k] += red[k + s];
        __syncthreads();
    }
    const float n_total = red[0];
    __syncthreads();

    // perplexity = exp(-sum p log(p + 1e-10)), p = cn / N
    const float p = cn / (float)N_ROWS;
    red[k] = -p * __logf(p + 1e-10f);
    __syncthreads();
    for (int s = 256; s > 0; s >>= 1) {
        if (k < s) red[k] += red[k + s];
        __syncthreads();
    }
    const float psum = red[0];

    const float smoothed = (avg_c + 1e-5f) / (n_total + KCODES * 1e-5f) * n_total;

    #pragma unroll 4
    for (int d = 0; d < DIM; ++d) {
        const float dwv = ws[WS_DW + d * KCODES + k];
        const float e   = ema_dw[d * KCODES + k];
        const float nhd = e - (e - dwv) * 0.01f;
        out[OUT_NEMB + d * KCODES + k] = (nhd / debias) / smoothed;
    }

    if (k == 0) {
        out[OUT_LOSS] = 0.25f * ws[WS_LOSS] / (float)NUMEL;   // COMMITMENT_COST
        out[OUT_PERP] = __expf(psum);
    }
}

// ---------------------------------------------------------------------------
extern "C" void kernel_launch(void* const* d_in, const int* in_sizes, int n_in,
                              void* d_out, int out_size, void* d_ws, size_t ws_size,
                              hipStream_t stream) {
    const float* x       = (const float*)d_in[0];
    const float* emb     = (const float*)d_in[1];
    const float* ema_c   = (const float*)d_in[2];
    const float* ema_dw  = (const float*)d_in[3];
    const int*   counter = (const int*)d_in[4];
    float* out = (float*)d_out;
    float* ws  = (float*)d_ws;

    vq_prep<<<256, 256, 0, stream>>>(emb, ws);
    vq_main<<<512, 256, 0, stream>>>(x, out, ws);    // 4096 waves x 32 rows
    vq_fin<<<1, 512, 0, stream>>>(ema_c, ema_dw, counter, out, ws);
}
